// RNNTLoss_51702816309747
// MI455X (gfx1250) — compile-verified, matrix operand order checked by reading, not run on hardware
//
#include <hip/hip_runtime.h>
#include <hip/hip_bf16.h>
#include <math.h>

// ---------------------------------------------------------------------------
// RNNT loss, gfx1250 (MI455X), wave32.
//
// Phase 1 (memory-bound, 662 MB @ 23.3 TB/s ~= 28 us):
//   per-(b,t,u) row of V=1024 logits -> logsumexp via
//   v_wmma_f32_16x16x32_f16 with an all-ones B matrix (row-sum on the matrix
//   pipe, f32 accumulation of f16 exp terms). 1 wave = 16 rows.
// Phase 2 (tiny): alpha forward recursion using the closed form
//   alpha_t = E + logcumsumexp(c - E), two associative block scans per t.
// ---------------------------------------------------------------------------

typedef __attribute__((ext_vector_type(16))) _Float16 v16h;
typedef __attribute__((ext_vector_type(8)))  float    v8f;

namespace {
constexpr int kB  = 8;
constexpr int kT  = 200;
constexpr int kU  = 100;
constexpr int kU1 = 101;
constexpr int kV  = 1024;
constexpr int kBlank = kV - 1;
constexpr long long kRows  = (long long)kB * kT * kU1;   // 161,600
constexpr int kTiles = (int)(kRows / 16);                // 10,100 (exact)
}

// ---------------------------------------------------------------------------
// Kernel 1: per-row logsumexp via WMMA row-sums; emit blank_lp / emit_lp.
// One wave (32 lanes) handles a 16-row tile. Lane L covers half of row
// (L & 15): float offsets c*32 + (L>>4)*16 .. +15 for chunk c in [0,32).
// ---------------------------------------------------------------------------
__global__ __launch_bounds__(256) void rnnt_lse_wmma_kernel(
    const float* __restrict__ logits,     // (B,T,U1,V)
    const int*   __restrict__ targets,    // (B,U)
    float* __restrict__ blank_lp,         // (B,T,U1)
    float* __restrict__ emit_lp)          // (B,T,U)
{
  const int wave = blockIdx.x * (blockDim.x >> 5) + (threadIdx.x >> 5);
  if (wave >= kTiles) return;             // wave-uniform guard: EXEC stays full

  const int lane = threadIdx.x & 31;
  const int m    = lane & 15;             // row within tile
  const int half = lane >> 4;             // which 16-wide half of each K chunk
  const long long row = (long long)wave * 16 + m;
  const float* __restrict__ rp = logits + row * kV;
  const float4* __restrict__ rp4 = (const float4*)rp;

  // ---- pass 1: row max over this lane's half; combine halves via shfl ----
  float mx = -INFINITY;
  for (int c = 0; c < 32; ++c) {
#pragma unroll
    for (int q = 0; q < 4; ++q) {
      float4 vv = rp4[c * 8 + half * 4 + q];
      mx = fmaxf(mx, fmaxf(fmaxf(vv.x, vv.y), fmaxf(vv.z, vv.w)));
    }
  }
  mx = fmaxf(mx, __shfl_xor(mx, 16, 32)); // both half-lanes now hold row max

  // ---- pass 2: sum(exp(x - mx)) via WMMA with all-ones B ----
  v16h ones;
#pragma unroll
  for (int j = 0; j < 16; ++j) ones[j] = (_Float16)1.0f;

  v8f acc = {};
  for (int c = 0; c < 32; ++c) {
    v16h a;
#pragma unroll
    for (int q = 0; q < 4; ++q) {
      float4 vv = rp4[c * 8 + half * 4 + q];
      a[4 * q + 0] = (_Float16)__expf(vv.x - mx);
      a[4 * q + 1] = (_Float16)__expf(vv.y - mx);
      a[4 * q + 2] = (_Float16)__expf(vv.z - mx);
      a[4 * q + 3] = (_Float16)__expf(vv.w - mx);
    }
    // D[m][n] += sum_k A[m,k] * 1  -> running f32 row sums
    acc = __builtin_amdgcn_wmma_f32_16x16x32_f16(
        /*neg_a=*/false, a, /*neg_b=*/false, ones,
        /*c_mod=*/(short)0, acc, /*reuse_a=*/false, /*reuse_b=*/false);
  }

  // C/D layout: VGPR j holds M=j on lanes 0-15 and M=8+j on lanes 16-31
  // (all columns equal since B==ones). Redistribute so lane m owns row m.
  float av[8];
#pragma unroll
  for (int j = 0; j < 8; ++j) av[j] = acc[j];
  float mine  = av[m & 7];                 // lanes 0-15: rows 0-7; 16-31: 8-15
  float other = __shfl_xor(mine, 16, 32);  // swap halves
  float rowsum = (m & 8) ? other : mine;   // valid for lanes 0-15

  if (lane < 16) {
    const float lse = mx + __logf(rowsum);
    const int        u  = (int)(row % kU1);
    const long long  bt = row / kU1;       // b*T + t
    const int        b  = (int)(bt / kT);
    blank_lp[row] = rp[kBlank] - lse;
    if (u < kU) {
      const int tgt = targets[b * kU + u];
      emit_lp[bt * kU + u] = rp[tgt] - lse;
    }
  }
}

// ---------------------------------------------------------------------------
// Kernel 2: alpha recursion. One 128-thread block per batch element.
// ---------------------------------------------------------------------------
__device__ __forceinline__ float logaddexpf_(float a, float b) {
  float m = fmaxf(a, b);
  if (m == -INFINITY) return -INFINITY;
  return m + __logf(__expf(a - m) + __expf(b - m));
}

__device__ __forceinline__ float block_incl_sum(float x, int lane, int wid,
                                                float* wred) {
#pragma unroll
  for (int o = 1; o < 32; o <<= 1) {
    float y = __shfl_up(x, o, 32);
    if (lane >= o) x += y;
  }
  if (lane == 31) wred[wid] = x;
  __syncthreads();
  float pre = 0.0f;
  for (int w = 0; w < wid; ++w) pre += wred[w];
  x += pre;
  __syncthreads();
  return x;
}

__device__ __forceinline__ float block_incl_lse(float x, int lane, int wid,
                                                float* wred) {
#pragma unroll
  for (int o = 1; o < 32; o <<= 1) {
    float y = __shfl_up(x, o, 32);
    if (lane >= o) x = logaddexpf_(x, y);
  }
  if (lane == 31) wred[wid] = x;
  __syncthreads();
  float pre = -INFINITY;
  for (int w = 0; w < wid; ++w) pre = logaddexpf_(pre, wred[w]);
  x = logaddexpf_(x, pre);
  __syncthreads();
  return x;
}

__global__ __launch_bounds__(128) void rnnt_alpha_kernel(
    const float* __restrict__ blank_lp,       // (B,T,U1)
    const float* __restrict__ emit_lp,        // (B,T,U)
    const int*   __restrict__ logit_lengths,  // (B,)
    const int*   __restrict__ target_lengths, // (B,)
    float* __restrict__ out)                  // (B,)
{
  const int b    = blockIdx.x;
  const int u    = threadIdx.x;      // 0..127, active for u < U1
  const int lane = u & 31;
  const int wid  = u >> 5;
  const bool act = (u < kU1);

  __shared__ float sAlpha[kU1];
  __shared__ float wsum[4];
  __shared__ float wlog[4];

  const int t_last = logit_lengths[b] - 1;
  const int u_fin  = target_lengths[b];
  const float* __restrict__ blp = blank_lp + (long long)b * kT * kU1;
  const float* __restrict__ elp = emit_lp  + (long long)b * kT * kU;

  // t = 0: alpha0[u] = sum_{k<u} emit[0][k] (exclusive prefix sum)
  float e0 = (act && u >= 1) ? elp[u - 1] : 0.0f;
  float alpha = block_incl_sum(e0, lane, wid, wsum);
  if (act) sAlpha[u] = alpha;
  __syncthreads();
  if (t_last == 0 && u == u_fin) out[b] = -(alpha + blp[u]);

  for (int t = 1; t < kT; ++t) {
    const float aprev = act ? sAlpha[u] : -INFINITY;
    const float c = act ? (aprev + blp[(t - 1) * kU1 + u]) : -INFINITY;
    const float e = (act && u >= 1) ? elp[t * kU + (u - 1)] : 0.0f;
    // E[u] = sum_{k<u} emit_t[k]; scans contain the barriers that protect
    // the sAlpha read above from the write below.
    const float E = block_incl_sum(e, lane, wid, wsum);
    const float z = c - E;                       // inactive lanes: -inf
    const float s = block_incl_lse(z, lane, wid, wlog);
    const float anew = E + s;
    if (act) sAlpha[u] = anew;
    __syncthreads();
    if (t == t_last && u == u_fin) out[b] = -(anew + blp[t * kU1 + u]);
  }
}

// ---------------------------------------------------------------------------
extern "C" void kernel_launch(void* const* d_in, const int* in_sizes, int n_in,
                              void* d_out, int out_size, void* d_ws,
                              size_t ws_size, hipStream_t stream) {
  (void)in_sizes; (void)n_in; (void)out_size; (void)ws_size;
  const float* logits  = (const float*)d_in[0];
  const int*   targets = (const int*)d_in[1];
  const int*   loglen  = (const int*)d_in[2];
  const int*   tgtlen  = (const int*)d_in[3];
  float* out = (float*)d_out;

  float* blank_lp = (float*)d_ws;                       // B*T*U1 floats
  float* emit_lp  = blank_lp + (size_t)kB * kT * kU1;   // B*T*U floats

  constexpr int waves_per_block = 8;                    // 256 threads, wave32
  const int blocks = (kTiles + waves_per_block - 1) / waves_per_block;
  rnnt_lse_wmma_kernel<<<blocks, 256, 0, stream>>>(logits, targets,
                                                   blank_lp, emit_lp);
  rnnt_alpha_kernel<<<kB, 128, 0, stream>>>(blank_lp, emit_lp, loglen, tgtlen,
                                            out);
}
